// NoMCOutModel_53128745452237
// MI455X (gfx1250) — matrix operand order repaired and checked by reading, not compiled, not run on hardware
//
#include <hip/hip_runtime.h>

typedef __attribute__((ext_vector_type(16))) _Float16 v16h;
typedef __attribute__((ext_vector_type(8)))  float    v8f;

#define B_    256
#define T_    256
#define M_    8
#define AUX_  32
#define H_    64
#define TC    16              // timesteps per chunk (== WMMA M)
#define NCHUNK (T_ / TC)      // 16
#define RCOLS (H_ * H_)       // 4096
#define JCOLS (M_ * H_)       // 512

// ---------------------------------------------------------------------------
// Pre-pass: convert f32 row-major weights [AUX, N] into f16 column-major
// [N][AUX] so each WMMA B-operand lane reads 16 contiguous f16 (32 bytes).
// ---------------------------------------------------------------------------
__global__ void convert_weights_kernel(const float* __restrict__ Wj,
                                       const float* __restrict__ Wr,
                                       _Float16* __restrict__ WjH,
                                       _Float16* __restrict__ WrH) {
  int i = blockIdx.x * blockDim.x + threadIdx.x;
  if (i < JCOLS * AUX_) {
    int n = i >> 5, k = i & 31;
    WjH[i] = (_Float16)Wj[k * JCOLS + n];
  }
  if (i < RCOLS * AUX_) {
    int n = i >> 5, k = i & 31;
    WrH[i] = (_Float16)Wr[k * RCOLS + n];
  }
}

// ---------------------------------------------------------------------------
// Main kernel: one workgroup per batch element, 8 waves, chunked time scan.
// ---------------------------------------------------------------------------
__global__ void nomc_scan_kernel(const float* __restrict__ x_m,
                                 const float* __restrict__ x_a,
                                 const float* __restrict__ bj,
                                 const float* __restrict__ br,
                                 const float* __restrict__ Wo,
                                 const float* __restrict__ bo,
                                 const float* __restrict__ Wfc,
                                 const float* __restrict__ bfc,
                                 const _Float16* __restrict__ WjH,
                                 const _Float16* __restrict__ WrH,
                                 float* __restrict__ out) {
  extern __shared__ char smem_raw[];
  float* sR   = (float*)smem_raw;            // [TC][4096] R logits -> softmax(r)
  float* sJ   = sR  + TC * RCOLS;            // [TC][512]  J logits -> softmax(j)
  float* sXa  = sJ  + TC * JCOLS;            // [TC][32]   xa chunk
  float* sXm  = sXa + TC * AUX_;             // [TC][8]    xm chunk
  float* sMin = sXm + TC * M_;               // [TC][64]   m_in
  float* sC   = sMin + TC * H_;              // [64]       carry c
  float* sA   = sC  + H_;                    // [64]       a = o * c (last step)

  const int b    = blockIdx.x;
  const int tid  = threadIdx.x;
  const int lane = tid & 31;
  const int wave = tid >> 5;

  if (tid < H_) sC[tid] = 0.f;

  // WMMA operand mapping (wave32, 16-bit); A and B both use contiguous k = 16*half + i
  const int arow = lane & 15;                // A: M = lane (mod 16)
  const int akb  = (lane < 16) ? 0 : 16;     // A: K base offset per half-wave
  const int bk0  = (lane < 16) ? 0 : 16;     // B: K base per half-wave
  const int bn   = lane & 15;                // B/C: N = lane (mod 16)
  const int crow = (lane < 16) ? 0 : 8;      // C: M base per half-wave

  for (int chunk = 0; chunk < NCHUNK; ++chunk) {
    const int t0 = chunk * TC;
    __syncthreads();
    // Stage xa / xm chunk (contiguous in global)
    for (int i = tid; i < TC * AUX_; i += blockDim.x)
      sXa[i] = x_a[((size_t)b * T_ + t0) * AUX_ + i];
    for (int i = tid; i < TC * M_; i += blockDim.x)
      sXm[i] = x_m[((size_t)b * T_ + t0) * M_ + i];
    __syncthreads();

    // Build A operand (16x32 f16), shared by every tile this chunk.
    v16h aA;
#pragma unroll
    for (int i = 0; i < 16; ++i) {
      int k = akb + i;
      aA[i] = (_Float16)sXa[arow * AUX_ + k];
    }

    // ---- R logits: 16x4096 = 256 tiles, one v_wmma_f32_16x16x32_f16 each ----
    for (int tile = wave; tile < RCOLS / 16; tile += 8) {
      const int c0 = tile * 16;
      const _Float16* bp = WrH + (size_t)(c0 + bn) * AUX_ + bk0;
      v16h bB;
#pragma unroll
      for (int i = 0; i < 16; ++i) bB[i] = bp[i];
      v8f acc = {};
      acc = __builtin_amdgcn_wmma_f32_16x16x32_f16(false, aA, false, bB,
                                                   (short)0, acc, false, false);
      const int ccol = c0 + bn;
      const float bias = br[ccol];
#pragma unroll
      for (int v = 0; v < 8; ++v)
        sR[(crow + v) * RCOLS + ccol] = acc[v] + bias;
    }

    // ---- J logits: 16x512 = 32 tiles ----
    for (int tile = wave; tile < JCOLS / 16; tile += 8) {
      const int c0 = tile * 16;
      const _Float16* bp = WjH + (size_t)(c0 + bn) * AUX_ + bk0;
      v16h bB;
#pragma unroll
      for (int i = 0; i < 16; ++i) bB[i] = bp[i];
      v8f acc = {};
      acc = __builtin_amdgcn_wmma_f32_16x16x32_f16(false, aA, false, bB,
                                                   (short)0, acc, false, false);
      const int ccol = c0 + bn;
      const float bias = bj[ccol];
#pragma unroll
      for (int v = 0; v < 8; ++v)
        sJ[(crow + v) * JCOLS + ccol] = acc[v] + bias;
    }
    __syncthreads();

    // ---- Softmax over contiguous 64-groups; skew index to dodge 32-way
    //      bank conflicts (all group bases are multiples of 64 floats). ----
    for (int g = tid; g < TC * H_; g += blockDim.x) {     // 1024 R groups
      float* p = sR + (g >> 6) * RCOLS + (g & 63) * H_;
      float mx = -1e30f;
      for (int i = 0; i < H_; ++i) mx = fmaxf(mx, p[(i + g) & 63]);
      float s = 0.f;
      for (int i = 0; i < H_; ++i) {
        int ix = (i + g) & 63;
        float e = __expf(p[ix] - mx);
        p[ix] = e; s += e;
      }
      float inv = 1.f / s;
      for (int i = 0; i < H_; ++i) p[(i + g) & 63] *= inv;
    }
    for (int g = tid; g < TC * M_; g += blockDim.x) {     // 128 J groups
      float* p = sJ + (g >> 3) * JCOLS + (g & 7) * H_;
      float mx = -1e30f;
      for (int i = 0; i < H_; ++i) mx = fmaxf(mx, p[(i + g) & 63]);
      float s = 0.f;
      for (int i = 0; i < H_; ++i) {
        int ix = (i + g) & 63;
        float e = __expf(p[ix] - mx);
        p[ix] = e; s += e;
      }
      float inv = 1.f / s;
      for (int i = 0; i < H_; ++i) p[(i + g) & 63] *= inv;
    }
    __syncthreads();

    // ---- m_in[t][h] = sum_m xm[t][m] * j[t][m][h] ----
    for (int i = tid; i < TC * H_; i += blockDim.x) {
      int t = i >> 6, h = i & 63;
      float s = 0.f;
#pragma unroll
      for (int m = 0; m < M_; ++m)
        s += sXm[t * M_ + m] * sJ[t * JCOLS + m * H_ + h];
      sMin[i] = s;
    }
    __syncthreads();

    // ---- Sequential 64-wide linear recurrence over the chunk ----
    for (int tt = 0; tt < TC; ++tt) {
      float nc = 0.f;
      if (tid < H_) {
        const float* rrow = sR + tt * RCOLS;
        float s = 0.f;
        for (int h = 0; h < H_; ++h) s += sC[h] * rrow[h * H_ + tid];
        nc = sMin[tt * H_ + tid] + s;
      }
      __syncthreads();
      if (tid < H_) sC[tid] = nc;
      __syncthreads();
    }
  }

  // ---- Epilogue: out gate at t = T-1, final FC, write outputs ----
  if (tid < H_) {
    const int h = tid;
    const float* xlast = sXa + (TC - 1) * AUX_;   // xa[b, T-1, :]
    float s = bo[h];
    for (int k = 0; k < AUX_; ++k) s += xlast[k] * Wo[k * H_ + h];
    float o = 1.f / (1.f + __expf(-s));
    sA[h] = o * sC[h];
    out[B_ + (size_t)b * H_ + h] = sC[h];         // c_final after out[0:256]
  }
  __syncthreads();
  if (tid == 0) {
    float s = bfc[0];
    for (int h = 0; h < H_; ++h) s += sA[h] * Wfc[h];
    out[b] = s;                                   // out[b, 0]
  }
}

// ---------------------------------------------------------------------------
extern "C" void kernel_launch(void* const* d_in, const int* in_sizes, int n_in,
                              void* d_out, int out_size, void* d_ws, size_t ws_size,
                              hipStream_t stream) {
  (void)in_sizes; (void)n_in; (void)out_size; (void)ws_size;
  const float* x_m = (const float*)d_in[0];
  const float* x_a = (const float*)d_in[1];
  const float* Wj  = (const float*)d_in[2];
  const float* bj  = (const float*)d_in[3];
  const float* Wr  = (const float*)d_in[4];
  const float* br  = (const float*)d_in[5];
  const float* Wo  = (const float*)d_in[6];
  const float* bo  = (const float*)d_in[7];
  const float* Wfc = (const float*)d_in[8];
  const float* bfc = (const float*)d_in[9];
  float* out = (float*)d_out;

  _Float16* WjH = (_Float16*)d_ws;                 // 512*32 f16 = 32 KB
  _Float16* WrH = WjH + (size_t)JCOLS * AUX_;      // 4096*32 f16 = 256 KB

  convert_weights_kernel<<<(RCOLS * AUX_ + 255) / 256, 256, 0, stream>>>(
      Wj, Wr, WjH, WrH);

  const size_t smem =
      (size_t)(TC * RCOLS + TC * JCOLS + TC * AUX_ + TC * M_ + TC * H_ + H_ + H_)
      * sizeof(float);                             // 302080 B < 320 KB WGP LDS
  nomc_scan_kernel<<<B_, 256, smem, stream>>>(
      x_m, x_a, bj, br, Wo, bo, Wfc, bfc, WjH, WrH, out);
}
